// MyGRUModel_38311108280444
// MI455X (gfx1250) — matrix operand order repaired and checked by reading, not compiled, not run on hardware
//
#include <hip/hip_runtime.h>

// Problem sizes (fixed by the reference)
#define T_SEQ   128
#define NBATCH  64
#define NHID    1024
#define NGATE   3072        // 3*NHID
#define NTOK    32000
#define NM      (T_SEQ * NBATCH)   // 8192 rows for the big GEMMs

typedef __attribute__((ext_vector_type(16))) __bf16 v16bf;
typedef __attribute__((ext_vector_type(8)))  __bf16 v8bf;
typedef __attribute__((ext_vector_type(8)))  float  v8f;

// Async global->LDS support detection (CDNA5 ASYNCcnt path)
#if defined(__has_builtin)
# if __has_builtin(__builtin_amdgcn_global_load_async_to_lds_b128)
#  define HAVE_ASYNC 1
# endif
#endif
#ifndef HAVE_ASYNC
# define HAVE_ASYNC 0
#endif

// Exact operand type per clang diagnostic: int __attribute__((vector_size(16)))
typedef int v4i_ __attribute__((vector_size(16)));
#if HAVE_ASYNC
typedef v4i_ __attribute__((address_space(1))) g_v4i;   // global (device) b128
typedef v4i_ __attribute__((address_space(3))) l_v4i;   // LDS b128
#endif

// LDS A-tile layout: 64 rows, chunk of 128 k-elements, padded stride
#define LDS_LDA 136   // elements; 272 B row stride (multiple of 16 B, bank-skewed)

// ---------------------------------------------------------------- helpers ---

__device__ __forceinline__ unsigned short f2bf(float f) {
  unsigned u = __float_as_uint(f);
  u += 0x7FFFu + ((u >> 16) & 1u);          // round-to-nearest-even
  return (unsigned short)(u >> 16);
}

// A fragment (16x32 bf16, row-major source [., ld]):
// lane L (<16): row M=L, K = kb+{0..7} in v0-3 and kb+{16..23} in v4-7
// lane L (>=16): row M=L-16, K = kb+{8..15} and kb+{24..31}
__device__ __forceinline__ v16bf load_a_frag(const unsigned short* __restrict__ A,
                                             int row, int ld, int kb, int hi) {
  union { v16bf v; v8bf h[2]; } u;
  const unsigned short* p = A + (size_t)row * ld + kb + hi * 8;
  u.h[0] = *(const v8bf*)p;         // 16B load
  u.h[1] = *(const v8bf*)(p + 16);  // 16B load
  return u.v;
}

// Same fragment but sourced from the LDS-staged chunk (row-major, LDS_LDA)
__device__ __forceinline__ v16bf load_a_frag_lds(const unsigned short* sA,
                                                 int row, int kk, int hi) {
  union { v16bf v; v8bf h[2]; } u;
  const unsigned short* p = sA + row * LDS_LDA + kk + hi * 8;
  u.h[0] = *(const v8bf*)p;         // ds_load 16B
  u.h[1] = *(const v8bf*)(p + 16);  // ds_load 16B
  return u.v;
}

// B fragment (32x16 bf16) from row-major W [N, K] (i.e. B = W^T):
// lane L: column N = L%16 == row L%16 of W; K = kb + (L/16)*8 + {0..7} and kb + 16 + (L/16)*8 + {0..7}
__device__ __forceinline__ v16bf load_b_frag(const unsigned short* __restrict__ W,
                                             int row, int ld, int kb, int hi) {
  union { v16bf v; v8bf h[2]; } u;
  const unsigned short* p = W + (size_t)row * ld + kb + hi * 8;
  u.h[0] = *(const v8bf*)p;         // 16B load
  u.h[1] = *(const v8bf*)(p + 16);  // 16B load
  return u.v;
}

__device__ __forceinline__ v8f wmma_bf16(v16bf a, v16bf b, v8f c) {
  // (neg_a, A, neg_b, B, c_mod, C, reuse_a, reuse_b)
  return __builtin_amdgcn_wmma_f32_16x16x32_bf16(false, a, false, b, (short)0, c,
                                                 false, false);
}

__device__ __forceinline__ float sigm(float x) { return 1.0f / (1.0f + __expf(-x)); }

// Stage one 64 x 128 bf16 A-chunk (contiguous 16 KB panel slice) into LDS.
// 256 threads: thread = (row<<2)|seg4, each moves 4 x 16B.
__device__ __forceinline__ void stage_chunk(const unsigned short* __restrict__ A,
                                            int mBlock, int K, int kb0,
                                            unsigned short* dst) {
  const int tid = threadIdx.x;
  const int r = tid >> 2;
  const int s0 = (tid & 3) * 4;               // first 16B segment (of 16 per row)
  const unsigned short* g = A + (size_t)(mBlock + r) * K + kb0 + s0 * 8;
  unsigned short* l = dst + r * LDS_LDA + s0 * 8;
  #pragma unroll
  for (int j = 0; j < 4; ++j) {
#if HAVE_ASYNC
    __builtin_amdgcn_global_load_async_to_lds_b128(
        (g_v4i*)(g + j * 8), (l_v4i*)(l + j * 8), 0, 0);
#else
    *(uint4*)(l + j * 8) = *(const uint4*)(g + j * 8);   // b128 load + ds_store
#endif
  }
}

__device__ __forceinline__ void wait_async_all() {
#if HAVE_ASYNC
# if defined(__has_builtin) && __has_builtin(__builtin_amdgcn_s_wait_asynccnt)
  __builtin_amdgcn_s_wait_asynccnt(0);
# else
  asm volatile("s_wait_asynccnt 0x0" ::: "memory");
# endif
#endif
}

// ------------------------------------------------------- conversion kernels ---

__global__ void __launch_bounds__(256)
cvt_f32_bf16(const float* __restrict__ src, unsigned short* __restrict__ dst, int n) {
  int i = blockIdx.x * 256 + threadIdx.x;
  if (i < n) dst[i] = f2bf(src[i]);
}

__global__ void __launch_bounds__(256)
embed_bf16(const int* __restrict__ tokens, const float* __restrict__ encW,
           unsigned short* __restrict__ x) {
  int m = blockIdx.x;                       // m = t*NBATCH + b
  int tok = tokens[m];
  const float* src = encW + (size_t)tok * NHID;
  unsigned short* dst = x + (size_t)m * NHID;
  for (int i = threadIdx.x; i < NHID; i += 256) dst[i] = f2bf(src[i]);
}

__global__ void __launch_bounds__(256)
hinit(const float* __restrict__ hsrc, float* __restrict__ h,
      unsigned short* __restrict__ hbf) {
  int i = blockIdx.x * 256 + threadIdx.x;   // NBATCH*NHID = 65536 elements
  float v = hsrc[i];
  h[i] = v;
  hbf[i] = f2bf(v);
}

// -------------------------------------------------- generic bf16 NT GEMM ---
// C[M,N] = A[M,K] * B[N,K]^T + bias[N], fp32 accumulate/out.
// Block: 256 threads = 8 waves as 2(M) x 4(N); wave tile 32x32; block 64x128.
// A-panel staged through LDS in double-buffered 64x128 chunks via async
// global->LDS copies (ASYNCcnt); B streams from L2 directly into WMMA regs.
__global__ void __launch_bounds__(256)
gemm_bf16nt(const unsigned short* __restrict__ A, const unsigned short* __restrict__ B,
            const float* __restrict__ bias, float* __restrict__ C,
            int M, int N, int K) {
  __shared__ unsigned short sA[2][64 * LDS_LDA];   // 2 x 17 KB

  const int w = threadIdx.x >> 5;
  const int lane = threadIdx.x & 31;
  const int laneLo = lane & 15, hi = lane >> 4;
  const int mBlock = blockIdx.x * 64;
  const int mLoc = (w & 1) * 32;                   // wave's row offset in block
  const int nBase = blockIdx.y * 128 + (w >> 1) * 32;
  const int bRow0 = nBase + laneLo, bRow1 = nBase + 16 + laneLo;

  v8f acc00 = {}, acc01 = {}, acc10 = {}, acc11 = {};

  const int nck = K >> 7;                          // K/128 chunks
  stage_chunk(A, mBlock, K, 0, sA[0]);
  for (int c = 0; c < nck; ++c) {
    wait_async_all();
    __syncthreads();                               // chunk c resident; prior reads done
    if (c + 1 < nck) stage_chunk(A, mBlock, K, (c + 1) << 7, sA[(c + 1) & 1]);
    const unsigned short* sa = sA[c & 1];
    const int kb0 = c << 7;
    if (kb0 + 256 < K) {
      __builtin_prefetch(B + (size_t)bRow0 * K + kb0 + 256, 0, 1);
      __builtin_prefetch(B + (size_t)bRow1 * K + kb0 + 256, 0, 1);
    }
    #pragma unroll
    for (int kk = 0; kk < 128; kk += 32) {
      v16bf a0 = load_a_frag_lds(sa, mLoc + laneLo, kk, hi);
      v16bf a1 = load_a_frag_lds(sa, mLoc + 16 + laneLo, kk, hi);
      v16bf b0 = load_b_frag(B, bRow0, K, kb0 + kk, hi);
      v16bf b1 = load_b_frag(B, bRow1, K, kb0 + kk, hi);
      acc00 = wmma_bf16(a0, b0, acc00);
      acc01 = wmma_bf16(a0, b1, acc01);
      acc10 = wmma_bf16(a1, b0, acc10);
      acc11 = wmma_bf16(a1, b1, acc11);
    }
  }

  // C/D layout: lane L vgpr v -> m = v + 8*(L/16), n = L%16
  #pragma unroll
  for (int mt = 0; mt < 2; ++mt) {
    #pragma unroll
    for (int nt = 0; nt < 2; ++nt) {
      v8f a = (mt == 0) ? (nt == 0 ? acc00 : acc01) : (nt == 0 ? acc10 : acc11);
      int col = nBase + nt * 16 + laneLo;
      float bv = bias[col];
      #pragma unroll
      for (int v = 0; v < 8; ++v) {
        int row = mBlock + mLoc + mt * 16 + hi * 8 + v;
        C[(size_t)row * N + col] = a[v] + bv;
      }
    }
  }
}

// ------------------------------------------------------- grid-wide barrier ---

__device__ __forceinline__ void grid_sync(unsigned* cnt, unsigned* gen, unsigned nblk) {
  __threadfence();            // make this thread's global stores device-visible
  __syncthreads();
  if (threadIdx.x == 0) {
    unsigned g = __hip_atomic_load(gen, __ATOMIC_RELAXED, __HIP_MEMORY_SCOPE_AGENT);
    unsigned a = __hip_atomic_fetch_add(cnt, 1u, __ATOMIC_ACQ_REL,
                                        __HIP_MEMORY_SCOPE_AGENT);
    if (a == nblk - 1) {
      __hip_atomic_store(cnt, 0u, __ATOMIC_RELAXED, __HIP_MEMORY_SCOPE_AGENT);
      __hip_atomic_store(gen, g + 1u, __ATOMIC_RELEASE, __HIP_MEMORY_SCOPE_AGENT);
    } else {
      while (__hip_atomic_load(gen, __ATOMIC_ACQUIRE, __HIP_MEMORY_SCOPE_AGENT) == g)
        __builtin_amdgcn_s_sleep(1);
    }
  }
  __syncthreads();
}

// ------------------------------------------------ persistent GRU recurrence ---
// One launch processes all T timesteps of one layer. Grid = 32 blocks.
// Per step: gh = h_bf16 * Whh^T + bhh  (each block: 96 of 3072 columns),
// barrier, fused GRU elementwise on 32 h-columns per block, barrier.
__global__ void __launch_bounds__(256)
gru_layer(const float* __restrict__ gi,           // [T][NBATCH][NGATE], bias included
          const unsigned short* __restrict__ Whh, // [NGATE][NHID] bf16
          const float* __restrict__ bhh,          // [NGATE]
          float* __restrict__ h,                  // [NBATCH][NHID] fp32 (in/out)
          unsigned short* __restrict__ hbf,       // [NBATCH][NHID] bf16 (in/out)
          unsigned short* __restrict__ xout,      // [T][NBATCH][NHID] bf16
          float* __restrict__ gh,                 // [NBATCH][NGATE] scratch
          unsigned* __restrict__ bar,             // {count, generation}
          int T) {
  const int bk = blockIdx.x;                 // 0..31
  const int w = threadIdx.x >> 5;
  const int lane = threadIdx.x & 31;
  const int laneLo = lane & 15, hi = lane >> 4;
  const int mt = w & 3;                      // M-tile 0..3 (NBATCH=64 rows)
  const int ng = w >> 2;                     // 0..1 -> 3 N-tiles each
  const int aRow = mt * 16 + laneLo;
  const int nb0 = bk * 96 + ng * 48;         // first gate column of this wave
  unsigned* cnt = bar;
  unsigned* gen = bar + 1;

  const int bRow0 = nb0 + laneLo;
  const int bRow1 = nb0 + 16 + laneLo;
  const int bRow2 = nb0 + 32 + laneLo;

  for (int t = 0; t < T; ++t) {
    // ---- gh = h * Whh^T (+bhh) for this block's 96 columns -----------------
    v8f acc0 = {}, acc1 = {}, acc2 = {};
    for (int kb = 0; kb < NHID; kb += 32) {
      v16bf af = load_a_frag(hbf, aRow, NHID, kb, hi);
      v16bf b0 = load_b_frag(Whh, bRow0, NHID, kb, hi);
      v16bf b1 = load_b_frag(Whh, bRow1, NHID, kb, hi);
      v16bf b2 = load_b_frag(Whh, bRow2, NHID, kb, hi);
      acc0 = wmma_bf16(af, b0, acc0);
      acc1 = wmma_bf16(af, b1, acc1);
      acc2 = wmma_bf16(af, b2, acc2);
    }
    #pragma unroll
    for (int j = 0; j < 3; ++j) {
      v8f a = (j == 0) ? acc0 : (j == 1 ? acc1 : acc2);
      int col = nb0 + j * 16 + laneLo;
      float bv = bhh[col];
      #pragma unroll
      for (int v = 0; v < 8; ++v) {
        int row = mt * 16 + hi * 8 + v;
        gh[row * NGATE + col] = a[v] + bv;
      }
    }
    grid_sync(cnt, gen, gridDim.x);

    // ---- fused GRU update: this block owns h columns [bk*32, bk*32+32) -----
    const float* giT = gi + (size_t)t * (NBATCH * NGATE);
    const int e0 = threadIdx.x * 8;          // 256 threads * 8 = 64x32 elements
    #pragma unroll
    for (int i = 0; i < 8; ++i) {
      int e = e0 + i;
      int b = e >> 5;                        // batch row 0..63
      int c = e & 31;
      int col = bk * 32 + c;                 // h column
      float i_r = giT[b * NGATE + col];
      float i_i = giT[b * NGATE + NHID + col];
      float i_n = giT[b * NGATE + 2 * NHID + col];
      float h_r = gh[b * NGATE + col];
      float h_i = gh[b * NGATE + NHID + col];
      float h_n = gh[b * NGATE + 2 * NHID + col];
      float hv  = h[b * NHID + col];
      float r = sigm(i_r + h_r);
      float z = sigm(i_i + h_i);
      float n = tanhf(i_n + r * h_n);
      float hy = n + z * (hv - n);
      h[b * NHID + col] = hy;
      unsigned short hb = f2bf(hy);
      hbf[b * NHID + col] = hb;
      xout[(size_t)t * (NBATCH * NHID) + b * NHID + col] = hb;
    }
    grid_sync(cnt, gen, gridDim.x);
  }
}

// ------------------------------------------------------------------ launch ---

extern "C" void kernel_launch(void* const* d_in, const int* in_sizes, int n_in,
                              void* d_out, int out_size, void* d_ws, size_t ws_size,
                              hipStream_t stream) {
  (void)in_sizes; (void)n_in; (void)out_size; (void)ws_size;

  const int*   tokens = (const int*)d_in[0];     // [T, B]
  const float* hidden = (const float*)d_in[1];   // [L, B, H]
  const float* encW   = (const float*)d_in[2];   // [V, H]
  const float* wih    = (const float*)d_in[3];   // [L, 3H, H]
  const float* whh    = (const float*)d_in[4];   // [L, 3H, H]
  const float* bih    = (const float*)d_in[5];   // [L, 3H]
  const float* bhh    = (const float*)d_in[6];   // [L, 3H]
  const float* decW   = (const float*)d_in[7];   // [V, H]
  const float* decb   = (const float*)d_in[8];   // [V]
  float* out = (float*)d_out;

  unsigned char* ws = (unsigned char*)d_ws;
  size_t off = 0;
  auto take = [&](size_t bytes) -> void* {
    void* p = ws + off;
    off += (bytes + 255) & ~(size_t)255;
    return p;
  };
  unsigned*       bar    = (unsigned*)take(256);
  unsigned short* wihb   = (unsigned short*)take((size_t)2 * NGATE * NHID * 2);
  unsigned short* whhb   = (unsigned short*)take((size_t)2 * NGATE * NHID * 2);
  unsigned short* decb16 = (unsigned short*)take((size_t)NTOK * NHID * 2);
  unsigned short* xA     = (unsigned short*)take((size_t)NM * NHID * 2); // emb / L1 out
  unsigned short* xB     = (unsigned short*)take((size_t)NM * NHID * 2); // L0 out
  float*          gi     = (float*)take((size_t)NM * NGATE * 4);
  float*          gh     = (float*)take((size_t)NBATCH * NGATE * 4);
  float*          h0     = (float*)take((size_t)NBATCH * NHID * 4);
  unsigned short* h0b    = (unsigned short*)take((size_t)NBATCH * NHID * 2);
  float*          h1     = (float*)take((size_t)NBATCH * NHID * 4);
  unsigned short* h1b    = (unsigned short*)take((size_t)NBATCH * NHID * 2);

  (void)hipMemsetAsync(bar, 0, 2 * sizeof(unsigned), stream);

  // Weight conversion fp32 -> bf16
  const int nW = 2 * NGATE * NHID;
  cvt_f32_bf16<<<(nW + 255) / 256, 256, 0, stream>>>(wih, wihb, nW);
  cvt_f32_bf16<<<(nW + 255) / 256, 256, 0, stream>>>(whh, whhb, nW);
  const int nD = NTOK * NHID;
  cvt_f32_bf16<<<(nD + 255) / 256, 256, 0, stream>>>(decW, decb16, nD);

  // Embedding gather -> bf16 x
  embed_bf16<<<NM, 256, 0, stream>>>(tokens, encW, xA);

  // ---- Layer 0 ----
  dim3 g1(NM / 64, NGATE / 128);
  gemm_bf16nt<<<g1, 256, 0, stream>>>(xA, wihb, bih, gi, NM, NGATE, NHID);
  hinit<<<(NBATCH * NHID) / 256, 256, 0, stream>>>(hidden, h0, h0b);
  gru_layer<<<32, 256, 0, stream>>>(gi, whhb, bhh, h0, h0b, xB, gh, bar, T_SEQ);

  // ---- Layer 1 ----
  gemm_bf16nt<<<g1, 256, 0, stream>>>(xB, wihb + (size_t)NGATE * NHID, bih + NGATE,
                                      gi, NM, NGATE, NHID);
  hinit<<<(NBATCH * NHID) / 256, 256, 0, stream>>>(hidden + NBATCH * NHID, h1, h1b);
  gru_layer<<<32, 256, 0, stream>>>(gi, whhb + (size_t)NGATE * NHID, bhh + NGATE,
                                    h1, h1b, xA, gh, bar, T_SEQ);

  // ---- Decoder: out[8192, 32000] = x2 * decW^T + dec_b ----
  dim3 g2(NM / 64, NTOK / 128);
  gemm_bf16nt<<<g2, 256, 0, stream>>>(xA, decb16, decb, out, NM, NTOK, NHID);

  // ---- next_hidden tail ----
  const size_t DEC = (size_t)NM * NTOK;
  (void)hipMemcpyAsync(out + DEC, h0, (size_t)NBATCH * NHID * sizeof(float),
                       hipMemcpyDeviceToDevice, stream);
  (void)hipMemcpyAsync(out + DEC + NBATCH * NHID, h1,
                       (size_t)NBATCH * NHID * sizeof(float),
                       hipMemcpyDeviceToDevice, stream);
}
